// RNNNet_27539330302204
// MI455X (gfx1250) — matrix operand order repaired and checked by reading, not compiled, not run on hardware
//
#include <hip/hip_runtime.h>

#define Bsz 128
#define Tsz 512
#define Isz 64
#define Hsz 1024
#define Osz 12

typedef __attribute__((ext_vector_type(16))) __bf16 v16bf;
typedef __attribute__((ext_vector_type(8)))  __bf16 v8bf;
typedef __attribute__((ext_vector_type(8)))  float  v8f;

__device__ __forceinline__ unsigned short f2bf(float f) {
    union { float f; unsigned u; } v; v.f = f;
    unsigned r = v.u + 0x7fffu + ((v.u >> 16) & 1u);   // round-to-nearest-even
    return (unsigned short)(r >> 16);
}
__device__ __forceinline__ float bf2f(unsigned short s) {
    union { unsigned u; float f; } v; v.u = ((unsigned)s) << 16;
    return v.f;
}

// ---------------------------------------------------------------------------
// Kernel 0: convert W_hh (f32 [H][H], row n = output neuron, col k = input h)
// to bf16 bit patterns. B operand of WMMA reads rows of this directly.
// ---------------------------------------------------------------------------
__global__ __launch_bounds__(256)
void convert_whh(const float* __restrict__ W, unsigned short* __restrict__ Wb, int n) {
    int i = blockIdx.x * blockDim.x + threadIdx.x;
    if (i < n) Wb[i] = f2bf(W[i]);
}

// ---------------------------------------------------------------------------
// Kernel 1 (launched once per timestep t):
//   h_t = tanh( x[:,t,:]·W_ihᵀ + b_ih + b_hh + h_{t-1}·W_hhᵀ )
// One wave computes one 16x16 tile of h_t.  8 btiles * 64 ntiles = 512 waves
// = 64 blocks * 8 waves.  Recurrence GEMM uses v_wmma_f32_16x16x32_bf16 with
// f32 accumulation; xp-init stays f32 VALU for precision.
// ---------------------------------------------------------------------------
__global__ __launch_bounds__(256)
void rnn_step(const float* __restrict__ x,            // [B,T,I] f32
              const float* __restrict__ W_ih,         // [H,I]  f32
              const float* __restrict__ b_ih,         // [H]
              const float* __restrict__ b_hh,         // [H]
              const unsigned short* __restrict__ Whh, // [H,H]  bf16 bits
              const unsigned short* __restrict__ hprev, // [B,H] bf16 bits or null (t==0)
              unsigned short* __restrict__ hout,      // [B,H]  bf16 bits (hs slice for t)
              int t) {
    const int lane  = threadIdx.x & 31;
    const int wave  = threadIdx.x >> 5;
    const int tile  = blockIdx.x * 8 + wave;  // 0..511
    const int btile = tile >> 6;              // 0..7   (b rows / 16)
    const int ntile = tile & 63;              // 0..63  (h cols / 16)
    const int nl    = lane & 15;
    const int hi    = lane >> 4;              // 0 or 1 (lane half)
    const int ncol  = ntile * 16 + nl;        // global output column (h')

    // ---- accumulator init: xp tile (f32 VALU) ----
    // C/D layout: VGPR r <-> row m = hi*8 + r, col = nl.
    v8f c;
    const float bias = b_ih[ncol] + b_hh[ncol];
    #pragma unroll
    for (int r = 0; r < 8; ++r) c[r] = bias;
    {
        const float* wrow = W_ih + ncol * Isz;
        const int m0 = btile * 16 + hi * 8;
        #pragma unroll 4
        for (int i = 0; i < Isz; ++i) {
            const float wv = wrow[i];
            #pragma unroll
            for (int r = 0; r < 8; ++r)
                c[r] = fmaf(x[((m0 + r) * Tsz + t) * Isz + i], wv, c[r]);
        }
    }

    // ---- recurrence: c += h_prev(16x1024) x W_hh^T(1024x16) via WMMA bf16 ----
    if (hprev) {
        const int arow  = btile * 16 + nl;     // A row (batch index), M = lane%16
        const int koffA = hi * 8;              // A lane K base within 32-chunk
        const int koffB = hi * 8;              // B lane K base within 32-chunk
        const unsigned short* aptr = hprev + arow * Hsz;
        const unsigned short* bptr = Whh + (size_t)ncol * Hsz; // B[k][n]=Whh[n][k]
        #pragma unroll 4
        for (int kk = 0; kk < Hsz; kk += 32) {
            v8bf a0 = *(const v8bf*)(aptr + kk + koffA);        // K = base+0..7
            v8bf a1 = *(const v8bf*)(aptr + kk + koffA + 16);   // K = base+16..23
            v16bf afrag = __builtin_shufflevector(a0, a1,
                0, 1, 2, 3, 4, 5, 6, 7, 8, 9, 10, 11, 12, 13, 14, 15);
            v8bf b0 = *(const v8bf*)(bptr + kk + koffB);        // K = base+0..7
            v8bf b1 = *(const v8bf*)(bptr + kk + koffB + 16);   // K = base+16..23
            v16bf bfrag = __builtin_shufflevector(b0, b1,
                0, 1, 2, 3, 4, 5, 6, 7, 8, 9, 10, 11, 12, 13, 14, 15);
            c = __builtin_amdgcn_wmma_f32_16x16x32_bf16(
                    false, afrag, false, bfrag, (short)0, c, false, false);
        }
    }

    // ---- tanh + store bf16 into hidden-state history ----
    #pragma unroll
    for (int r = 0; r < 8; ++r) {
        const int b = btile * 16 + hi * 8 + r;
        hout[b * Hsz + ncol] = f2bf(tanhf(c[r]));
    }
}

// ---------------------------------------------------------------------------
// Kernel 2: out[b,t,o] = sum_h hs[t,b,h]*W_out[o,h] + b_out[o]
// One wave per (t,b) row; deterministic shuffle-tree reduction.
// ---------------------------------------------------------------------------
__global__ __launch_bounds__(256)
void out_proj(const unsigned short* __restrict__ hs, // [T,B,H] bf16 bits
              const float* __restrict__ Wout,        // [O,H]
              const float* __restrict__ bout,        // [O]
              float* __restrict__ out) {             // [B,T,O]
    const int lane = threadIdx.x & 31;
    const int wid  = (blockIdx.x * blockDim.x + threadIdx.x) >> 5; // 0..T*B-1
    const int t = wid >> 7;        // / B
    const int b = wid & (Bsz - 1); // % B
    const unsigned short* hrow = hs + (size_t)(t * Bsz + b) * Hsz;

    float acc[Osz];
    #pragma unroll
    for (int o = 0; o < Osz; ++o) acc[o] = 0.f;

    for (int h = lane; h < Hsz; h += 32) {
        const float hv = bf2f(hrow[h]);
        #pragma unroll
        for (int o = 0; o < Osz; ++o)
            acc[o] = fmaf(hv, Wout[o * Hsz + h], acc[o]);
    }
    #pragma unroll
    for (int o = 0; o < Osz; ++o) {
        float v = acc[o];
        #pragma unroll
        for (int off = 16; off > 0; off >>= 1)
            v += __shfl_down(v, off, 32);
        if (lane == 0)
            out[((size_t)b * Tsz + t) * Osz + o] = v + bout[o];
    }
}

// ---------------------------------------------------------------------------
extern "C" void kernel_launch(void* const* d_in, const int* in_sizes, int n_in,
                              void* d_out, int out_size, void* d_ws, size_t ws_size,
                              hipStream_t stream) {
    const float* x     = (const float*)d_in[0];
    const float* W_ih  = (const float*)d_in[1];
    const float* W_hh  = (const float*)d_in[2];
    const float* b_ih  = (const float*)d_in[3];
    const float* b_hh  = (const float*)d_in[4];
    const float* W_out = (const float*)d_in[5];
    const float* b_out = (const float*)d_in[6];
    float* out = (float*)d_out;

    // workspace layout: [W_hh bf16: 2MB][hs bf16: T*B*H*2 = 134MB]
    unsigned short* Whh_b = (unsigned short*)d_ws;
    unsigned short* hs    = Whh_b + (size_t)Hsz * Hsz;

    convert_whh<<<(Hsz * Hsz + 255) / 256, 256, 0, stream>>>(W_hh, Whh_b, Hsz * Hsz);

    for (int t = 0; t < Tsz; ++t) {
        const unsigned short* hp =
            (t == 0) ? nullptr : hs + (size_t)(t - 1) * Bsz * Hsz;
        rnn_step<<<64, 256, 0, stream>>>(x, W_ih, b_ih, b_hh, Whh_b, hp,
                                         hs + (size_t)t * Bsz * Hsz, t);
    }

    const int nwaves = Tsz * Bsz;              // 65536 rows
    out_proj<<<nwaves / 8, 256, 0, stream>>>(hs, W_out, b_out, out);
}
